// RegionProposalNetwork_87110526698174
// MI455X (gfx1250) — matrix operand order repaired and checked by reading, not compiled, not run on hardware
//
#include <hip/hip_runtime.h>
#include <hip/hip_bf16.h>

typedef __attribute__((ext_vector_type(16))) __bf16 v16bf;
typedef __attribute__((ext_vector_type(8)))  float  v8f;

#define NIMG   2
#define CCH    256
#define AN     3
#define LVTOP  600
#define IMTOP  300
#define NLVL   5
#define NALL   (NIMG * NLVL * LVTOP)   // 6000
#define SORTN  8192
#define BBOX_CLIP 4.135166556742356f
#define IOU_THR   0.7f
#define MIN_SIZE  1e-2f
#define GROUP_OFF 4096.0f

__device__ __forceinline__ unsigned short f2bf(float f) {
  unsigned u = __float_as_uint(f);
  u += 0x7FFFu + ((u >> 16) & 1u);     // round-to-nearest-even
  return (unsigned short)(u >> 16);
}
__device__ __forceinline__ float bf2f(unsigned short s) {
  return __uint_as_float(((unsigned)s) << 16);
}
// order-preserving uint key: larger float -> larger key
__device__ __forceinline__ unsigned fkey(float f) {
  unsigned u = __float_as_uint(f);
  return (u & 0x80000000u) ? ~u : (u | 0x80000000u);
}
// 0-based LDS byte address of a __shared__ object (for async-to-LDS vdst)
__device__ __forceinline__ unsigned ldsAddr(void* p) {
  return (unsigned)(size_t)(__attribute__((address_space(3))) void*)p;
}

// ---------------------------------------------------------------------------
// Prep: conv weights OIHW -> bf16 in FRAGMENT-MAJOR order:
//   wBs[kpos][cb][nt][lane][e]  (ushort), so a B fragment for (ntile, lane)
//   is 16 contiguous bf16 = 32 bytes -> 2x ds_load_b128.
//   Element semantics: co = nt*16 + (lane&15); k = e + 16*(lane>>4);
//                      ci = cb*32 + k.
// Also pack head weights [15][256] (3 logit + 12 reg rows) + bias[15].
// ---------------------------------------------------------------------------
__global__ __launch_bounds__(256) void prep_kernel(
    const float* __restrict__ conv_w, const float* __restrict__ log_w,
    const float* __restrict__ log_b, const float* __restrict__ reg_w,
    const float* __restrict__ reg_b, unsigned short* __restrict__ wBs,
    float* __restrict__ headW, float* __restrict__ headB)
{
  int i = blockIdx.x * 256 + threadIdx.x;
  if (i < 2304 * 256) {
    int e    = i & 15;
    int L    = (i >> 4) & 31;
    int nt   = (i >> 9) & 15;
    int cb   = (i >> 13) & 7;
    int kpos = i >> 16;
    int co = nt * 16 + (L & 15);
    int k  = e + 16 * (L >> 4);
    int ci = cb * 32 + k;
    int ky = kpos / 3, kx = kpos % 3;
    float v = conv_w[(((size_t)co * 256 + ci) * 3 + ky) * 3 + kx];
    wBs[i] = f2bf(v);
  }
  if (i < 16 * 256) {
    int j = i >> 8, c = i & 255;
    float v = 0.0f;
    if (j < 3) v = log_w[j * 256 + c];
    else if (j < 15) v = reg_w[(j - 3) * 256 + c];
    headW[i] = v;
    if (c == 0) headB[j] = (j < 3) ? log_b[j] : ((j < 15) ? reg_b[j - 3] : 0.0f);
  }
}

// ---------------------------------------------------------------------------
// Per-level fmap transpose + bf16 convert: NCHW f32 -> [n][pixel][ci] bf16.
// Tiled through LDS so both global read and write are coalesced/vectorized.
// ---------------------------------------------------------------------------
__global__ __launch_bounds__(256) void xpose_kernel(
    const float* __restrict__ fm, unsigned short* __restrict__ xTl, int HW)
{
  const int n = blockIdx.y;
  const int base = blockIdx.x * 64;
  const int tid = threadIdx.x;
  __shared__ __align__(16) unsigned short T[64 * 256];
  const float* fmn = fm + (size_t)n * CCH * HW;
  for (int idx = tid; idx < 64 * 256; idx += 256) {
    int ci = idx >> 6, m = idx & 63;          // consecutive tid -> consecutive p
    int p = base + m;
    float v = (p < HW) ? fmn[(size_t)ci * HW + p] : 0.0f;
    T[m * 256 + ci] = f2bf(v);
  }
  __syncthreads();
  uint4* out4 = (uint4*)(xTl + (size_t)n * HW * 256);
  const uint4* T4 = (const uint4*)T;
  for (int idx = tid; idx < 2048; idx += 256) { // 64 rows x 32 uint4
    int m = idx >> 5, part = idx & 31;
    int p = base + m;
    if (p < HW) out4[(size_t)p * 32 + part] = T4[m * 32 + part];
  }
}

// ---------------------------------------------------------------------------
// Fused 3x3 conv (implicit GEMM, bf16 WMMA) + ReLU + 15-ch head.
// Block: 256 threads = 8 waves. Tile: M=64 pixels, N=256 out channels.
// A tile (64x256 bf16, one kpos shift) staged once per kpos into AH;
// B blocks staged by GLOBAL_LOAD_ASYNC_TO_LDS_B128 into a double buffer,
// overlapped with WMMA on the other buffer. AH is reused as the H tile.
// ---------------------------------------------------------------------------
__global__ __launch_bounds__(256) void conv_head_kernel(
    const unsigned short* __restrict__ xT,   // [n][HW][256] bf16 (level base)
    const unsigned short* __restrict__ wBs,  // swizzled weights
    const float* __restrict__ bias, const float* __restrict__ headW,
    const float* __restrict__ headB, float* __restrict__ lbuf_l,
    float* __restrict__ rbuf_l, int H, int W)
{
  const int HW = H * W;
  const int n = blockIdx.y;
  const int base = blockIdx.x * 64;
  const int tid = threadIdx.x;
  const int lane = tid & 31, wv = tid >> 5;
  const int mb = wv & 3, nh = wv >> 2;

  __shared__ __align__(32) unsigned short AH[64 * 256];   // A tile, then H tile
  __shared__ __align__(32) unsigned short Blds[2][8192];  // double buffer

  v8f acc[8];
#pragma unroll
  for (int t = 0; t < 8; ++t) acc[t] = (v8f){0, 0, 0, 0, 0, 0, 0, 0};

  // async B staging: each thread moves 4x16B per block
  const unsigned myOff = (unsigned)tid * 16u;
  const unsigned ldsB0 = ldsAddr(&Blds[0][0]) + myOff;

  // A staging: 4 threads per pixel row, 8 uint4 (128 B) each
  const int am = tid >> 2;
  const int apart = (tid & 3) * 8;
  const int p_a = base + am;
  const int ya = p_a / W, xa = p_a % W;
  const unsigned short* xTn = xT + (size_t)n * HW * 256;

  const int arow = 16 * mb + (lane & 15);
  const int ahi = (lane >> 4) * 16;
  const int boff0 = (nh * 8) * 512 + lane * 16;  // ushort offset of nb=0 frag

  // prologue: kick off block 0 into buffer 0
  {
    unsigned go = myOff;
    unsigned lo = ldsB0;
#pragma unroll
    for (int q = 0; q < 4; ++q)
      asm volatile("global_load_async_to_lds_b128 %0, %1, %2"
                   :: "v"(lo + q * 4096u), "v"(go + q * 4096u), "s"(wBs)
                   : "memory");
  }

  for (int kpos = 0; kpos < 9; ++kpos) {
    const int ky = kpos / 3 - 1, kx = kpos % 3 - 1;
    const int ys = ya + ky, xs = xa + kx;
    const bool okp = (p_a < HW) && (ys >= 0) && (ys < H) && (xs >= 0) && (xs < W);
    const long ps = (long)ys * W + xs;
    const uint4* asrc = (const uint4*)(xTn + ps * 256);
    uint4* adst = (uint4*)(&AH[am * 256]);
    const uint4 zz = make_uint4(0, 0, 0, 0);
#pragma unroll
    for (int q = 0; q < 8; ++q)
      adst[apart + q] = okp ? asrc[apart + q] : zz;

    for (int cb = 0; cb < 8; ++cb) {
      const int step = kpos * 8 + cb;
      const int cur = cb & 1;                 // == step & 1
      if (step < 71) {
        // issue next block into the other buffer (overlaps with compute)
        unsigned go = (unsigned)(step + 1) * 16384u + myOff;
        unsigned lo = ldsB0 + (unsigned)(cur ^ 1) * 16384u;
#pragma unroll
        for (int q = 0; q < 4; ++q)
          asm volatile("global_load_async_to_lds_b128 %0, %1, %2"
                       :: "v"(lo + q * 4096u), "v"(go + q * 4096u), "s"(wBs)
                       : "memory");
        // async loads complete in order: <=4 outstanding => current buf ready
        asm volatile("s_wait_asynccnt 0x4" ::: "memory");
      } else {
        asm volatile("s_wait_asynccnt 0x0" ::: "memory");
      }
      __syncthreads();

      // A fragment: two contiguous 16 B runs -> 2x ds_load_b128
      v16bf af;
      {
        const unsigned short* ap = &AH[arow * 256 + cb * 32 + ahi];
        ((uint4*)&af)[0] = *(const uint4*)(ap);
        ((uint4*)&af)[1] = *(const uint4*)(ap + 8);
      }
      const unsigned short* bbase = &Blds[cur][0];
#pragma unroll
      for (int nb = 0; nb < 8; ++nb) {
        // B fragment: 32 contiguous bytes per lane -> 2x ds_load_b128
        v16bf bf;
        const unsigned short* bp = bbase + boff0 + nb * 512;
        ((uint4*)&bf)[0] = ((const uint4*)bp)[0];
        ((uint4*)&bf)[1] = ((const uint4*)bp)[1];
        acc[nb] = __builtin_amdgcn_wmma_f32_16x16x32_bf16(
            false, af, false, bf, (short)0, acc[nb], false, false);
      }
      __syncthreads();   // no wave may overwrite a buffer others still read
    }
  }

  // bias + ReLU -> AH (reused as H tile).  D layout: VGPR r, M = r+8*(lane>=16)
#pragma unroll
  for (int nb = 0; nb < 8; ++nb) {
#pragma unroll
    for (int r = 0; r < 8; ++r) {
      int row = 16 * mb + r + 8 * (lane >> 4);
      int col = nh * 128 + nb * 16 + (lane & 15);
      float v = acc[nb][r] + bias[col];
      v = v > 0.0f ? v : 0.0f;
      AH[row * 256 + col] = f2bf(v);
    }
  }
  __syncthreads();

  // fused head: 64 pixels x 15 outputs, dot over 256 channels
  float* lout = lbuf_l + (size_t)n * HW * AN;
  float* rout = rbuf_l + (size_t)n * HW * AN * 4;
  for (int i = tid; i < 64 * 15; i += 256) {
    int m = i / 15, j = i % 15;
    int p = base + m;
    if (p >= HW) continue;
    float s = headB[j];
    const float* hw = headW + j * 256;
#pragma unroll 8
    for (int c = 0; c < 256; ++c) s += bf2f(AH[m * 256 + c]) * hw[c];
    if (j < 3) {
      lout[p * 3 + j] = s;
    } else {
      int jr = j - 3;
      rout[(p * 3 + (jr >> 2)) * 4 + (jr & 3)] = s;
    }
  }
}

// ---------------------------------------------------------------------------
// Per-(image,level) top-600 via 4-pass radix select on float keys, then
// compaction + decode + clip + validity into the candidate arrays.
// Candidate slot = n*3000 + l*600 + rank (matches reference concat order).
// ---------------------------------------------------------------------------
__global__ __launch_bounds__(256) void topk_kernel(
    const float* __restrict__ lbuf, const float* __restrict__ rbuf,
    const float* p0, const float* p1, const float* p2, const float* p3,
    const float* p4, const float* __restrict__ imsizes,
    float* __restrict__ cbox, float* __restrict__ cscore,
    int* __restrict__ cvalid, int* __restrict__ cgroup)
{
  const int l = blockIdx.x, n = blockIdx.y, tid = threadIdx.x;
  const int HWs[5] = {60800, 15200, 3800, 950, 247};
  const long LOFF[5] = {0, 364800, 456000, 478800, 484500};
  const int NLV = HWs[l] * 3;
  const float* lg = lbuf + LOFF[l] + (long)n * NLV;
  const float* rg = rbuf + 4 * (LOFF[l] + (long)n * NLV);
  const float* pris[5] = {p0, p1, p2, p3, p4};
  const float* pri = pris[l];

  __shared__ unsigned hist[256];
  __shared__ unsigned s_prefix, s_mask, s_remaining, s_cg, s_c1, s_c2;
  if (tid == 0) { s_prefix = 0; s_mask = 0; s_remaining = LVTOP; s_cg = 0; }
  __syncthreads();

  for (int shift = 24; shift >= 0; shift -= 8) {
    if (tid < 256) hist[tid] = 0;
    __syncthreads();
    unsigned pref = s_prefix, msk = s_mask;
    for (int i = tid; i < NLV; i += 256) {
      unsigned k = fkey(lg[i]);
      if ((k & msk) == pref) atomicAdd(&hist[(k >> shift) & 255], 1u);
    }
    __syncthreads();
    if (tid == 0) {
      unsigned cum = 0;
      int b = 255;
      for (; b > 0; --b) {
        unsigned h = hist[b];
        if (cum + h >= s_remaining) break;
        cum += h;
      }
      s_cg += cum;
      s_remaining -= cum;
      s_prefix |= ((unsigned)b) << shift;
      s_mask |= 255u << shift;
    }
    __syncthreads();
  }
  const unsigned T = s_prefix;
  const unsigned cg = s_cg;
  if (tid == 0) { s_c1 = 0; s_c2 = 0; }
  __syncthreads();

  const float imh = imsizes[n * 2 + 0];
  const float imw = imsizes[n * 2 + 1];
  const long gbase = (long)n * (NLVL * LVTOP) + (long)l * LVTOP;

  for (int i = tid; i < NLV; i += 256) {
    float f = lg[i];
    unsigned k = fkey(f);
    int slot = -1;
    if (k > T) slot = (int)atomicAdd(&s_c1, 1u);
    else if (k == T) {
      unsigned q = atomicAdd(&s_c2, 1u);
      if (cg + q < LVTOP) slot = (int)(cg + q);
    }
    if (slot < 0) continue;
    // decode
    float px1 = pri[i * 4 + 0], py1 = pri[i * 4 + 1];
    float px2 = pri[i * 4 + 2], py2 = pri[i * 4 + 3];
    float pw = px2 - px1, ph = py2 - py1;
    float pcx = px1 + 0.5f * pw, pcy = py1 + 0.5f * ph;
    float dx = rg[i * 4 + 0], dy = rg[i * 4 + 1];
    float dw = fminf(rg[i * 4 + 2], BBOX_CLIP);
    float dh = fminf(rg[i * 4 + 3], BBOX_CLIP);
    float cx = dx * pw + pcx, cy = dy * ph + pcy;
    float w = pw * __expf(dw), h = ph * __expf(dh);
    float x1 = cx - 0.5f * w, y1 = cy - 0.5f * h;
    float x2 = cx + 0.5f * w, y2 = cy + 0.5f * h;
    // clip
    x1 = fminf(fmaxf(x1, 0.0f), imw);
    y1 = fminf(fmaxf(y1, 0.0f), imh);
    x2 = fminf(fmaxf(x2, 0.0f), imw);
    y2 = fminf(fmaxf(y2, 0.0f), imh);
    int vld = ((x2 - x1) >= MIN_SIZE) && ((y2 - y1) >= MIN_SIZE);
    float sc = 1.0f / (1.0f + __expf(-f));
    long g = gbase + slot;
    cbox[g * 4 + 0] = x1; cbox[g * 4 + 1] = y1;
    cbox[g * 4 + 2] = x2; cbox[g * 4 + 3] = y2;
    cscore[g] = sc;
    cvalid[g] = vld;
    cgroup[g] = n * 10 + l;
  }
}

// ---------------------------------------------------------------------------
// Global descending score sort (stable on index) of 6000 candidates.
// Bitonic over 8192 packed keys: (~scorekey)<<32 | idx, ascending.
// ---------------------------------------------------------------------------
__global__ __launch_bounds__(1024) void sort_kernel(
    const float* __restrict__ cscore, const int* __restrict__ cvalid,
    int* __restrict__ sidx)
{
  __shared__ unsigned long long keys[SORTN];
  const int tid = threadIdx.x;
  for (int i = tid; i < SORTN; i += 1024) {
    unsigned sk = 0;  // pads: smallest key -> end
    if (i < NALL) {
      float s = cvalid[i] ? cscore[i] : -1.0f;
      sk = fkey(s);
    }
    keys[i] = ((unsigned long long)(~sk) << 32) | (unsigned)i;
  }
  __syncthreads();
  for (unsigned k = 2; k <= SORTN; k <<= 1) {
    for (unsigned j = k >> 1; j > 0; j >>= 1) {
      for (unsigned i = tid; i < SORTN; i += 1024) {
        unsigned ij = i ^ j;
        if (ij > i) {
          bool up = ((i & k) == 0);
          unsigned long long a = keys[i], b = keys[ij];
          if ((a > b) == up) { keys[i] = b; keys[ij] = a; }
        }
      }
      __syncthreads();
    }
  }
  for (int i = tid; i < NALL; i += 1024)
    sidx[i] = (int)(keys[i] & 0xFFFFFFFFull);
}

// ---------------------------------------------------------------------------
// Greedy batched NMS over sorted order. GROUP_OFF trick => cross-group IoU=0.
// ---------------------------------------------------------------------------
__global__ __launch_bounds__(1024) void nms_kernel(
    const float* __restrict__ cbox, const int* __restrict__ cvalid,
    const int* __restrict__ cgroup, const int* __restrict__ sidx,
    float* __restrict__ sbox, float* __restrict__ sarea,
    int* __restrict__ skeep)
{
  __shared__ unsigned char keep[NALL];
  const int tid = threadIdx.x;
  for (int i = tid; i < NALL; i += 1024) {
    int j = sidx[i];
    float off = (float)cgroup[j] * GROUP_OFF;
    float x1 = cbox[j * 4 + 0] + off, y1 = cbox[j * 4 + 1] + off;
    float x2 = cbox[j * 4 + 2] + off, y2 = cbox[j * 4 + 3] + off;
    sbox[i * 4 + 0] = x1; sbox[i * 4 + 1] = y1;
    sbox[i * 4 + 2] = x2; sbox[i * 4 + 3] = y2;
    sarea[i] = (x2 - x1) * (y2 - y1);
    keep[i] = (unsigned char)(cvalid[j] != 0);
  }
  __syncthreads();
  for (int i = 0; i < NALL; ++i) {
    if (keep[i]) {
      float x1 = sbox[i * 4 + 0], y1 = sbox[i * 4 + 1];
      float x2 = sbox[i * 4 + 2], y2 = sbox[i * 4 + 3];
      float ai = sarea[i];
      for (int jj = i + 1 + tid; jj < NALL; jj += 1024) {
        if (!keep[jj]) continue;
        float bx1 = sbox[jj * 4 + 0], by1 = sbox[jj * 4 + 1];
        float bx2 = sbox[jj * 4 + 2], by2 = sbox[jj * 4 + 3];
        float iw = fminf(x2, bx2) - fmaxf(x1, bx1);
        float ih = fminf(y2, by2) - fmaxf(y1, by1);
        iw = fmaxf(iw, 0.0f); ih = fmaxf(ih, 0.0f);
        float inter = iw * ih;
        float iou = inter / (ai + sarea[jj] - inter);
        if (iou > IOU_THR) keep[jj] = 0;
      }
    }
    __syncthreads();
  }
  for (int i = tid; i < NALL; i += 1024) skeep[i] = keep[i];
}

// ---------------------------------------------------------------------------
// Per-image cap at 300, write d_out = [bout 2400 | imout 600 | vout 600].
// ---------------------------------------------------------------------------
__global__ __launch_bounds__(256) void out_kernel(
    const float* __restrict__ cbox, const int* __restrict__ sidx,
    const int* __restrict__ skeep, float* __restrict__ dout)
{
  const int tid = threadIdx.x;
  for (int i = tid; i < 3600; i += 256) dout[i] = 0.0f;
  for (int t = tid; t < NIMG * IMTOP; t += 256)
    dout[2400 + t] = (float)(t / IMTOP);   // imout
  __syncthreads();
  if (tid == 0) {
    int cnt[NIMG] = {0, 0};
    for (int i = 0; i < NALL; ++i) {
      if (!skeep[i]) continue;
      int j = sidx[i];
      int im = j / (NLVL * LVTOP);
      if (cnt[im] < IMTOP) {
        int slot = im * IMTOP + cnt[im];
        dout[slot * 4 + 0] = cbox[j * 4 + 0];
        dout[slot * 4 + 1] = cbox[j * 4 + 1];
        dout[slot * 4 + 2] = cbox[j * 4 + 2];
        dout[slot * 4 + 3] = cbox[j * 4 + 3];
        dout[3000 + slot] = 1.0f;           // vout
        cnt[im]++;
      }
    }
  }
}

// ---------------------------------------------------------------------------
extern "C" void kernel_launch(void* const* d_in, const int* in_sizes, int n_in,
                              void* d_out, int out_size, void* d_ws, size_t ws_size,
                              hipStream_t stream) {
  (void)in_sizes; (void)n_in; (void)out_size; (void)ws_size;
  const float* fmaps[5]  = {(const float*)d_in[0], (const float*)d_in[1],
                            (const float*)d_in[2], (const float*)d_in[3],
                            (const float*)d_in[4]};
  const float* priors[5] = {(const float*)d_in[5], (const float*)d_in[6],
                            (const float*)d_in[7], (const float*)d_in[8],
                            (const float*)d_in[9]};
  const float* imsizes = (const float*)d_in[10];
  const float* conv_w  = (const float*)d_in[11];
  const float* conv_b  = (const float*)d_in[12];
  const float* log_w   = (const float*)d_in[13];
  const float* log_b   = (const float*)d_in[14];
  const float* reg_w   = (const float*)d_in[15];
  const float* reg_b   = (const float*)d_in[16];
  float* dout = (float*)d_out;

  static const int h_H[5]  = {200, 100, 50, 25, 13};
  static const int h_W[5]  = {304, 152, 76, 38, 19};
  static const int h_HW[5] = {60800, 15200, 3800, 950, 247};
  static const long h_LOFF[5] = {0, 364800, 456000, 478800, 484500};
  static const long h_CUMHW[5] = {0, 60800, 76000, 79800, 80750};
  const long LTOT = 485982;          // total logit floats
  const long XTOT = 41470464;        // 2 imgs * 80997 px * 256 ch (bf16)

  // workspace layout
  char* ws = (char*)d_ws;
  size_t off = 0;
  auto take = [&](size_t bytes) {
    char* p = ws + off;
    off = (off + bytes + 255) & ~(size_t)255;
    return p;
  };
  unsigned short* wBs = (unsigned short*)take((size_t)2304 * 256 * 2);
  float* headW  = (float*)take(16 * 256 * 4);
  float* headB  = (float*)take(16 * 4);
  float* lbuf   = (float*)take((size_t)LTOT * 4);
  float* rbuf   = (float*)take((size_t)LTOT * 4 * 4);
  float* cbox   = (float*)take((size_t)NALL * 4 * 4);
  float* cscore = (float*)take((size_t)NALL * 4);
  int*   cvalid = (int*)take((size_t)NALL * 4);
  int*   cgroup = (int*)take((size_t)NALL * 4);
  int*   sidx   = (int*)take((size_t)NALL * 4);
  int*   skeep  = (int*)take((size_t)NALL * 4);
  float* sbox   = (float*)take((size_t)NALL * 4 * 4);
  float* sarea  = (float*)take((size_t)NALL * 4);
  unsigned short* xT = (unsigned short*)take((size_t)XTOT * 2);

  prep_kernel<<<2304, 256, 0, stream>>>(conv_w, log_w, log_b, reg_w, reg_b,
                                        wBs, headW, headB);
  for (int l = 0; l < 5; ++l) {
    dim3 grid((h_HW[l] + 63) / 64, NIMG);
    xpose_kernel<<<grid, 256, 0, stream>>>(fmaps[l], xT + h_CUMHW[l] * 512,
                                           h_HW[l]);
  }
  for (int l = 0; l < 5; ++l) {
    dim3 grid((h_HW[l] + 63) / 64, NIMG);
    conv_head_kernel<<<grid, 256, 0, stream>>>(
        xT + h_CUMHW[l] * 512, wBs, conv_b, headW, headB,
        lbuf + h_LOFF[l], rbuf + 4 * h_LOFF[l], h_H[l], h_W[l]);
  }
  {
    dim3 grid(NLVL, NIMG);
    topk_kernel<<<grid, 256, 0, stream>>>(
        lbuf, rbuf, priors[0], priors[1], priors[2], priors[3], priors[4],
        imsizes, cbox, cscore, cvalid, cgroup);
  }
  sort_kernel<<<1, 1024, 0, stream>>>(cscore, cvalid, sidx);
  nms_kernel<<<1, 1024, 0, stream>>>(cbox, cvalid, cgroup, sidx,
                                     sbox, sarea, skeep);
  out_kernel<<<1, 256, 0, stream>>>(cbox, sidx, skeep, dout);
}
